// MultiHeadAttention_50792283242690
// MI455X (gfx1250) — matrix operand order repaired and checked by reading, not compiled, not run on hardware
//
#include <hip/hip_runtime.h>
#include <hip/hip_bf16.h>

// Problem constants (B, Q, T, M, H, K, V) = (4, 2048, 2048, 1024, 16, 64, 64)
constexpr int Bv = 4;
constexpr int Qv = 2048;
constexpr int Tv = 2048;
constexpr int Mv = 1024;
constexpr int Hv = 16;
constexpr int Kv = 64;   // head dim (K == V == 64)

typedef __attribute__((ext_vector_type(16))) _Float16 v16h;
typedef __attribute__((ext_vector_type(8)))  float    v8f;
typedef __attribute__((ext_vector_type(4)))  float    f32x4;

// ---------------------------------------------------------------------------
// WMMA helper (V_WMMA_F32_16X16X32_F16, wave32)
// ---------------------------------------------------------------------------
__device__ __forceinline__ v8f wmma(v16h a, v16h b, v8f c) {
  return __builtin_amdgcn_wmma_f32_16x16x32_f16(
      /*neg_a=*/false, a, /*neg_b=*/false, b,
      /*c_mod=*/(short)0, c, /*reuse_a=*/false, /*reuse_b=*/false);
}

// A-matrix 16x32 fragment (ISA 7.12.2): lane holds row (lane&15);
// halves 0..7 -> K = kb..kb+7, halves 8..15 -> K = kb+16..kb+23, kb = 8*(lane>=16).
__device__ __forceinline__ v16h frag_ld_a(const _Float16* base, int stride) {
  const int lane = threadIdx.x & 31;
  const int r  = lane & 15;
  const int kb = (lane >> 4) << 3;
  const _Float16* p = base + (size_t)r * stride + kb;
  union { v16h v; f32x4 q[2]; } u;
  u.q[0] = *reinterpret_cast<const f32x4*>(p);
  u.q[1] = *reinterpret_cast<const f32x4*>(p + 16);
  return u.v;
}

// B-matrix 32x16 fragment: lane holds column (lane&15);
// halves 0..7 -> K = koff..koff+7, halves 8..15 -> K = koff+16..koff+23, koff = 8*(lane>=16).
__device__ __forceinline__ v16h frag_ld_b(const _Float16* base, int stride) {
  const int lane = threadIdx.x & 31;
  const int n    = lane & 15;
  const int koff = (lane >> 4) << 3;
  const _Float16* p = base + (size_t)n * stride + koff;
  union { v16h v; f32x4 q[2]; } u;
  u.q[0] = *reinterpret_cast<const f32x4*>(p);
  u.q[1] = *reinterpret_cast<const f32x4*>(p + 16);
  return u.v;
}

// Async memory->LDS copy, 16 bytes (CDNA5 GLOBAL_LOAD_ASYNC_TO_LDS_B128).
__device__ __forceinline__ void async_copy_b128(unsigned lds_off, int goff,
                                                const void* sbase) {
  asm volatile("global_load_async_to_lds_b128 %0, %1, %2"
               :: "v"(lds_off), "v"(goff), "s"(sbase) : "memory");
}
#define ASYNC_WAIT(n) asm volatile("s_wait_asynccnt " #n ::: "memory")

// ---------------------------------------------------------------------------
// Kernel 1: projection  out = in(b,r,:) x w(h,:,:)  (f32 -> f16)
// transpose_out=0: out[b,h,r,k];  transpose_out=1: out[b,h,k,r] (for async V).
// Register-prefetch pipeline: next global tiles load during current WMMAs.
// ---------------------------------------------------------------------------
__global__ __launch_bounds__(256) void proj_kernel(
    const float* __restrict__ in, const float* __restrict__ w,
    _Float16* __restrict__ outp, int rows, int transpose_out) {
  __shared__ _Float16 As[128][40];
  __shared__ _Float16 Bs[64][40];

  const int tid = threadIdx.x, lane = tid & 31, wave = tid >> 5;
  const int b = blockIdx.z, h = blockIdx.y, r0 = blockIdx.x * 128;
  const float* inb = in + (size_t)b * rows * Mv + (size_t)r0 * Mv;
  const float* wh  = w  + (size_t)h * Mv * Kv;

  float ar[16], br[8];
  auto loadA = [&](int m0) {
#pragma unroll
    for (int i = 0; i < 16; ++i) {
      int idx = tid + i * 256, r = idx >> 5, c = idx & 31;
      ar[i] = inb[(size_t)r * Mv + m0 + c];
    }
  };
  auto loadB = [&](int m0) {
#pragma unroll
    for (int i = 0; i < 8; ++i) {
      int idx = tid + i * 256, mm = idx >> 6, k = idx & 63;
      br[i] = wh[(size_t)(m0 + mm) * Kv + k];
    }
  };

  loadA(0); loadB(0);
  v8f acc[4] = {};
  for (int m0 = 0; m0 < Mv; m0 += 32) {
    __syncthreads();                       // previous compute reads done
#pragma unroll
    for (int i = 0; i < 16; ++i) {
      int idx = tid + i * 256, r = idx >> 5, c = idx & 31;
      As[r][c] = (_Float16)ar[i];
    }
#pragma unroll
    for (int i = 0; i < 8; ++i) {
      int idx = tid + i * 256, mm = idx >> 6, k = idx & 63;
      Bs[k][mm] = (_Float16)br[i];
    }
    if (m0 + 32 < Mv) { loadA(m0 + 32); loadB(m0 + 32); }  // overlap w/ WMMA
    __syncthreads();
    // Hoist all fragments first so ds_load latency hides under the WMMA burst.
    v16h a = frag_ld_a(&As[wave * 16][0], 40);
    v16h bf[4];
#pragma unroll
    for (int c = 0; c < 4; ++c) bf[c] = frag_ld_b(&Bs[c * 16][0], 40);
#pragma unroll
    for (int c = 0; c < 4; ++c) acc[c] = wmma(a, bf[c], acc[c]);
  }

  const int half8 = (lane >= 16) ? 8 : 0;
  if (transpose_out) {
    // out[((b*H+h)*64 + col) * rows + row]; 8 consecutive rows per lane -> 16B.
    _Float16* ob = outp + (size_t)(b * Hv + h) * Kv * rows;
    const int row0 = r0 + wave * 16 + half8;
#pragma unroll
    for (int c = 0; c < 4; ++c) {
      const int col = c * 16 + (lane & 15);
      union { f32x4 q; _Float16 hh[8]; } u;
#pragma unroll
      for (int j = 0; j < 8; ++j) u.hh[j] = (_Float16)acc[c][j];
      *reinterpret_cast<f32x4*>(ob + (size_t)col * rows + row0) = u.q;
    }
  } else {
    _Float16* ob = outp + ((size_t)(b * Hv + h) * rows + r0 + wave * 16) * Kv;
#pragma unroll
    for (int c = 0; c < 4; ++c)
#pragma unroll
      for (int j = 0; j < 8; ++j)
        ob[(size_t)(j + half8) * Kv + c * 16 + (lane & 15)] = (_Float16)acc[c][j];
  }
}

// ---------------------------------------------------------------------------
// Kernel 2: flash attention, double-buffered async K/V pipeline.
// LDS arena:  K0 @0  V0 @9216 | K1 @18432  V1 @27648 | Ps @36864 (8x2304)
// Tile = 4 async b128 instructions/wave -> s_wait_asynccnt 4 retires tile i
// while tile i+1's DMA overlaps the 16 WMMAs + softmax.
// ---------------------------------------------------------------------------
constexpr unsigned AKS0 = 0,     AVS0 = 9216;
constexpr unsigned AKS1 = 18432, AVS1 = 27648;
constexpr unsigned APS  = 36864;
constexpr unsigned ATTN_LDS = 55296;

__global__ __launch_bounds__(256) void attn_kernel(
    const _Float16* __restrict__ qp, const _Float16* __restrict__ kp,
    const _Float16* __restrict__ vp, const float* __restrict__ tmask,
    const float* __restrict__ qtmask, _Float16* __restrict__ pre) {
  extern __shared__ char smem[];

  const int tid = threadIdx.x, lane = tid & 31, wave = tid >> 5;
  _Float16* Psw = (_Float16*)(smem + APS + wave * 2304);
  const int b = blockIdx.z, h = blockIdx.y, q0 = blockIdx.x * 128;
  const size_t bh = (size_t)b * Hv + h;
  const int half8 = (lane >= 16) ? 8 : 0;
  const int qrow0 = q0 + wave * 16 + half8;

  auto issue_tile = [&](int t0, unsigned kOff, unsigned vOff) {
    const _Float16* kbp = kp + (bh * Tv + t0) * Kv;   // [t][k] rows of 128B
    const _Float16* vbp = vp + bh * Kv * Tv + t0;     // [d][t] rows, stride 2T
#pragma unroll
    for (int i = 0; i < 2; ++i) {
      int idx = tid + i * 256;
      int row = idx >> 3, ch = idx & 7;
      async_copy_b128(kOff + row * 144u + ch * 16u, row * 128 + ch * 16, kbp);
      async_copy_b128(vOff + row * 144u + ch * 16u, row * (Tv * 2) + ch * 16, vbp);
    }
  };

  // Q fragments live in registers for the whole T sweep.
  const _Float16* qrow = qp + (bh * Qv + q0 + wave * 16) * Kv;
  v16h qa0 = frag_ld_a(qrow, Kv);
  v16h qa1 = frag_ld_a(qrow + 32, Kv);

  v8f o[4] = {};
  float m_i[8], l_i[8];
#pragma unroll
  for (int j = 0; j < 8; ++j) { m_i[j] = -1e30f; l_i[j] = 0.f; }

  issue_tile(0, AKS0, AVS0);
  constexpr int NIT = Tv / 64;
  for (int it = 0; it < NIT; ++it) {
    const int t0 = it * 64;
    const unsigned kOff = (it & 1) ? AKS1 : AKS0;
    const unsigned vOff = (it & 1) ? AVS1 : AVS0;
    if (it + 1 < NIT) {
      issue_tile(t0 + 64, (it & 1) ? AKS0 : AKS1, (it & 1) ? AVS0 : AVS1);
      if (t0 + 128 < Tv)
        __builtin_prefetch(kp + (bh * Tv + t0 + 128) * Kv, 0, 1);
      ASYNC_WAIT(4);          // tile `it` (oldest 4 async ops) has landed
    } else {
      ASYNC_WAIT(0);
    }
    __syncthreads();          // all waves' portions of tile `it` arrived
    const _Float16* Ks = (const _Float16*)(smem + kOff);
    const _Float16* Vs = (const _Float16*)(smem + vOff);

    // S = Q * K^T : hoist K fragments per k-chunk, then WMMA bursts.
    v8f s[4] = {};
#pragma unroll
    for (int c2 = 0; c2 < 2; ++c2) {
      v16h kf[4];
#pragma unroll
      for (int c = 0; c < 4; ++c)
        kf[c] = frag_ld_b(Ks + (c * 16) * 72 + c2 * 32, 72);
      v16h qa = (c2 == 0) ? qa0 : qa1;
#pragma unroll
      for (int c = 0; c < 4; ++c) s[c] = wmma(qa, kf[c], s[c]);
    }

    // Mask + 1/sqrt(K) scale in C-layout.
    float sv[4][8];
#pragma unroll
    for (int c = 0; c < 4; ++c) {
      const int t_idx = t0 + c * 16 + (lane & 15);
      const float tm = tmask[(size_t)b * Tv + t_idx];
#pragma unroll
      for (int j = 0; j < 8; ++j) {
        float qtm = qtmask[(size_t)(qrow0 + j) * Tv + t_idx];
        float adj = fmaxf(qtm, tm) * -1.0e6f;
        sv[c][j] = (s[c][j] + adj) * 0.125f;
      }
    }

    // Online softmax per row (slot j within a 16-lane half-wave).
#pragma unroll
    for (int j = 0; j < 8; ++j) {
      float rm = fmaxf(fmaxf(sv[0][j], sv[1][j]), fmaxf(sv[2][j], sv[3][j]));
#pragma unroll
      for (int off = 8; off > 0; off >>= 1) rm = fmaxf(rm, __shfl_xor(rm, off, 16));
      float mn  = fmaxf(m_i[j], rm);
      float fac = __expf(m_i[j] - mn);
      float ls  = 0.f;
#pragma unroll
      for (int c = 0; c < 4; ++c) {
        float pe = __expf(sv[c][j] - mn);
        sv[c][j] = pe;
        ls += pe;
      }
#pragma unroll
      for (int off = 8; off > 0; off >>= 1) ls += __shfl_xor(ls, off, 16);
      m_i[j] = mn;
      l_i[j] = l_i[j] * fac + ls;
#pragma unroll
      for (int c = 0; c < 4; ++c) o[c][j] *= fac;
    }

    // P (C-layout) -> per-wave LDS strip -> A-layout fragments.
#pragma unroll
    for (int c = 0; c < 4; ++c)
#pragma unroll
      for (int j = 0; j < 8; ++j)
        Psw[(j + half8) * 72 + c * 16 + (lane & 15)] = (_Float16)sv[c][j];
    __syncthreads();

    // O += P * V : hoist V fragments per k-chunk.
#pragma unroll
    for (int c2 = 0; c2 < 2; ++c2) {
      v16h pa = frag_ld_a(Psw + c2 * 32, 72);
      v16h vf[4];
#pragma unroll
      for (int c = 0; c < 4; ++c)
        vf[c] = frag_ld_b(Vs + (c * 16) * 72 + c2 * 32, 72);
#pragma unroll
      for (int c = 0; c < 4; ++c) o[c] = wmma(pa, vf[c], o[c]);
    }
    __syncthreads();          // this buffer free before it+1 re-fills it
  }

  // pre[b, q, h*64 + d] = O / l
  _Float16* pb = pre + ((size_t)b * Qv + qrow0) * (size_t)(Hv * Kv) + h * Kv;
#pragma unroll
  for (int j = 0; j < 8; ++j) {
    float inv = l_i[j] > 0.f ? 1.0f / l_i[j] : 0.f;
#pragma unroll
    for (int c = 0; c < 4; ++c)
      pb[(size_t)j * (Hv * Kv) + c * 16 + (lane & 15)] = (_Float16)(o[c][j] * inv);
  }
}

// ---------------------------------------------------------------------------
// Kernel 3: output projection  out[b,q,m] = (pre[b,q,:] . wo[:,m]) * (1-qmask)
// Double-buffered async A-tiles + register-prefetched B-tiles.
// LDS: A0 @0 (10240) | A1 @10240 | B0 @20480 (5120) | B1 @25600
// ---------------------------------------------------------------------------
constexpr unsigned OAS0 = 0,     OAS1 = 10240;
constexpr unsigned OBS0 = 20480, OBS1 = 25600;
constexpr unsigned OP_LDS = 30720;

__global__ __launch_bounds__(256) void oproj_kernel(
    const _Float16* __restrict__ pre, const float* __restrict__ wo,
    const float* __restrict__ qmask, float* __restrict__ outp) {
  extern __shared__ char smem[];

  const int tid = threadIdx.x, lane = tid & 31, wave = tid >> 5;
  const int b = blockIdx.z, m0 = blockIdx.y * 64, q0 = blockIdx.x * 128;
  const _Float16* pb = pre + ((size_t)b * Qv + q0) * Mv;

  auto issueA = [&](int hd0, unsigned aOff) {
#pragma unroll
    for (int i = 0; i < 2; ++i) {
      int idx = tid + i * 256;
      int r = idx >> 2, ch = idx & 3;
      async_copy_b128(aOff + r * 80u + ch * 16u,
                      r * (Mv * 2) + hd0 * 2 + ch * 16, pb);
    }
  };
  float br[8];
  auto loadB = [&](int hd0) {
#pragma unroll
    for (int i = 0; i < 8; ++i) {
      int idx = tid + i * 256, hd = idx >> 6, mm = idx & 63;
      br[i] = wo[(size_t)(hd0 + hd) * Mv + m0 + mm];
    }
  };

  issueA(0, OAS0);
  loadB(0);
  v8f acc[4] = {};
  constexpr int NIT = Mv / 32;
  for (int it = 0; it < NIT; ++it) {
    const int hd0 = it * 32;
    const unsigned aOff = (it & 1) ? OAS1 : OAS0;
    _Float16* Bsb = (_Float16*)(smem + ((it & 1) ? OBS1 : OBS0));
#pragma unroll
    for (int i = 0; i < 8; ++i) {       // commit B regs (f32->f16, transposed)
      int idx = tid + i * 256, hd = idx >> 6, mm = idx & 63;
      Bsb[mm * 40 + hd] = (_Float16)br[i];
    }
    if (it + 1 < NIT) {
      issueA(hd0 + 32, (it & 1) ? OAS0 : OAS1);
      loadB(hd0 + 32);                  // overlaps with WMMAs below
      ASYNC_WAIT(4);                    // 2 ops/tile: A-tile `it` may still be in flight
    } else {
      ASYNC_WAIT(0);
    }
    __syncthreads();
    const _Float16* As = (const _Float16*)(smem + aOff);
    v16h a = frag_ld_a(As + (wave * 16) * 40, 40);
    v16h bf[4];
#pragma unroll
    for (int c = 0; c < 4; ++c) bf[c] = frag_ld_b(Bsb + (c * 16) * 40, 40);
#pragma unroll
    for (int c = 0; c < 4; ++c) acc[c] = wmma(a, bf[c], acc[c]);
    __syncthreads();                    // buffers free before refill
  }

  const int half8 = (lane >= 16) ? 8 : 0;
#pragma unroll
  for (int j = 0; j < 8; ++j) {
    int q = q0 + wave * 16 + j + half8;
    float sc = 1.0f - qmask[(size_t)b * Qv + q];
#pragma unroll
    for (int c = 0; c < 4; ++c)
      outp[((size_t)b * Qv + q) * Mv + m0 + c * 16 + (lane & 15)] = acc[c][j] * sc;
  }
}

// ---------------------------------------------------------------------------
// Host launcher
// ---------------------------------------------------------------------------
extern "C" void kernel_launch(void* const* d_in, const int* in_sizes, int n_in,
                              void* d_out, int out_size, void* d_ws, size_t ws_size,
                              hipStream_t stream) {
  (void)in_sizes; (void)n_in; (void)out_size; (void)ws_size;
  const float* kvinput = (const float*)d_in[0];
  const float* qinput  = (const float*)d_in[1];
  const float* qmask   = (const float*)d_in[2];
  const float* tmask   = (const float*)d_in[3];
  const float* qtmask  = (const float*)d_in[4];
  const float* wq      = (const float*)d_in[5];
  const float* wk      = (const float*)d_in[6];
  const float* wv      = (const float*)d_in[7];
  const float* wo      = (const float*)d_in[8];
  float* outp = (float*)d_out;

  // f16 workspace: qproj (B,H,Q,64) | kproj (B,H,T,64) | vprojT (B,H,64,T) | pre (B,Q,1024)
  const size_t per = (size_t)Bv * Hv * Qv * Kv;   // 8,388,608 halves = 16 MB
  _Float16* qp   = (_Float16*)d_ws;
  _Float16* kp   = qp + per;
  _Float16* vp   = kp + per;
  _Float16* prew = vp + per;

  dim3 blk(256);
  proj_kernel<<<dim3(Qv / 128, Hv, Bv), blk, 0, stream>>>(qinput,  wq, qp, Qv, 0);
  proj_kernel<<<dim3(Tv / 128, Hv, Bv), blk, 0, stream>>>(kvinput, wk, kp, Tv, 0);
  proj_kernel<<<dim3(Tv / 128, Hv, Bv), blk, 0, stream>>>(kvinput, wv, vp, Tv, 1);
  attn_kernel<<<dim3(Qv / 128, Hv, Bv), blk, ATTN_LDS, stream>>>(
      qp, kp, vp, tmask, qtmask, prew);
  oproj_kernel<<<dim3(Qv / 128, Mv / 64, Bv), blk, OP_LDS, stream>>>(
      prew, wo, qmask, outp);
}
